// fwd_gnn_45174466019865
// MI455X (gfx1250) — matrix operand order repaired and checked by reading, not compiled, not run on hardware
//
#include <hip/hip_runtime.h>
#include <hip/hip_bf16.h>

typedef __attribute__((ext_vector_type(16))) __bf16 v16bf;
typedef __attribute__((ext_vector_type(8)))  float  v8f;
typedef float        f32x4 __attribute__((ext_vector_type(4)));
typedef unsigned int u32x4 __attribute__((ext_vector_type(4)));
typedef int          i32x8 __attribute__((ext_vector_type(8)));
typedef int          i32x4 __attribute__((ext_vector_type(4)));

union Frag16 { v16bf v; uint4 q[2]; };
union B8     { uint4 q; __bf16 h[8]; };

static constexpr int kH    = 128;
static constexpr int kFEAT = 32;
static constexpr int kM    = 32768;
static constexpr int kL    = 16;
static constexpr int kMU   = kM / 2;
static constexpr long kN   = (long)kM * kL;

static constexpr int ROWS = 64;    // rows per block
static constexpr int AST  = 264;   // input LDS stride (256-wide + 4-dword pad)
static constexpr int WST1 = 136;   // 128-wide + 4-dword pad

#if __has_builtin(__builtin_amdgcn_tensor_load_to_lds) && \
    __has_builtin(__builtin_amdgcn_s_wait_tensorcnt)
#define HAVE_TDM 1
#else
#define HAVE_TDM 0
#endif

// ---------------------------------------------------------------------------
// fp32 -> bf16 weight conversion
// ---------------------------------------------------------------------------
__global__ void cvt_f32_bf16(const float* __restrict__ x, __bf16* __restrict__ y, int n) {
  int i = blockIdx.x * blockDim.x + threadIdx.x;
  if (i < n) y[i] = (__bf16)x[i];
}

// ---------------------------------------------------------------------------
// TDM: DMA one 128 x K bf16 weight tile (row-major) global -> LDS, inserting
// 4 DWORDs of pad per row so LDS stride becomes K+8 bf16 (matches compute).
// D# per CDNA5 ISA: group0 = {flags, lds_addr,
// global_addr, type=2}; group1 = {data_size=2B, pad_enable, pad_interval
// (64 or 128 dwords), pad_amount(4 dwords), dims/strides}; groups 2-3 and the
// trailing group are zero (2D tensor). Tracked by TENSORcnt.
// ---------------------------------------------------------------------------
#if HAVE_TDM
static __device__ inline void tdm_weight_load(const __bf16* wg, unsigned int ldsOff, int K) {
  unsigned long long ga = (unsigned long long)(uintptr_t)wg;
  u32x4 g0;
  g0.x = 0x1u;                                            // count=1, user mode
  g0.y = ldsOff;                                          // lds_addr (bytes)
  g0.z = (unsigned int)ga;                                // global_addr[31:0]
  g0.w = ((unsigned int)(ga >> 32) & 0x1FFFFFFu) | (2u << 30); // ga[56:32], type=2
  const int padInt = (K == 256) ? 6 : 5;                  // 128 / 64 dwords per row
  i32x8 g1;
  g1[0] = (1 << 16) | (1 << 20) | (padInt << 22) | (3 << 25); // data_size=2B, pad_en, pad=4dw
  g1[1] = (K & 0xFFFF) << 16;                             // tensor_dim0 lo16 @bit48
  g1[2] = (K >> 16) | (128 << 16);                        // tensor_dim0 hi, tensor_dim1=128
  g1[3] = (K & 0xFFFF) << 16;                             // tile_dim0 = K @bit112
  g1[4] = 128;                                            // tile_dim1 = 128 rows
  g1[5] = K;                                              // tensor_dim0_stride = K
  g1[6] = 0;
  g1[7] = 0;
  i32x4 z4 = {0, 0, 0, 0};
  i32x8 z8 = {0, 0, 0, 0, 0, 0, 0, 0};
  __builtin_amdgcn_tensor_load_to_lds(g0, g1, z4, z4, z8, 0);
}
#endif

// fallback: cooperative copy global bf16 (128 x K) -> LDS padded
static __device__ inline void load_weights(const __bf16* __restrict__ wg, __bf16* wl,
                                           int K, int tid) {
  const int csh   = (K == 256) ? 5 : 4;
  const int total = 128 << csh;
  const int strd  = K + 8;
  for (int idx = tid; idx < total; idx += 256) {
    int r = idx >> csh;
    int c = idx & ((1 << csh) - 1);
    *(uint4*)(wl + r * strd + c * 8) = *(const uint4*)(wg + r * K + c * 8);
  }
}

// ---------------------------------------------------------------------------
// One dense layer: out(64x128) = relu( in(64xK) @ W^T + b ), WMMA bf16.
// 8 waves: wave w -> row-tile (w>>1), col-tiles (w&1)*4 .. +3.
// A (ISA 7.12.2): lane=M; lanes<16 hold K{0..7,16..23}, lanes>=16 K{8..15,24..31}.
// B: lane=N; same K order as A, two 8-element runs from W row.
// ---------------------------------------------------------------------------
static __device__ inline void mlp_layer(const __bf16* __restrict__ inL, int inStride, int K,
                                        const __bf16* __restrict__ wL, int wStride,
                                        const float* __restrict__ biasG,
                                        __bf16* __restrict__ outL, int tid) {
  const int w = tid >> 5, lane = tid & 31;
  const int rt = w >> 1, ct0 = (w & 1) * 4;
  const int lm = lane & 15, hi = lane >> 4;
  v8f acc[4] = {};
  const __bf16* arow = inL + (rt * 16 + lm) * inStride;
  for (int k0 = 0; k0 < K; k0 += 32) {
    Frag16 a;
    a.q[0] = *(const uint4*)(arow + k0 + hi * 8);
    a.q[1] = *(const uint4*)(arow + k0 + 16 + hi * 8);
#pragma unroll
    for (int j = 0; j < 4; ++j) {
      const __bf16* wrow = wL + ((ct0 + j) * 16 + lm) * wStride + k0 + hi * 8;
      Frag16 b;
      b.q[0] = *(const uint4*)(wrow);
      b.q[1] = *(const uint4*)(wrow + 16);
      acc[j] = __builtin_amdgcn_wmma_f32_16x16x32_bf16(false, a.v, false, b.v,
                                                       (short)0, acc[j], false, false);
    }
  }
#pragma unroll
  for (int j = 0; j < 4; ++j) {
    const int col = (ct0 + j) * 16 + lm;
    const float bv = biasG[col];
#pragma unroll
    for (int r = 0; r < 8; ++r) {
      float v = acc[j][r] + bv;
      v = v > 0.f ? v : 0.f;
      outL[(rt * 16 + hi * 8 + r) * WST1 + col] = (__bf16)v;
    }
  }
}

// residual: o = a + b  (64 x 128, stride WST1)
static __device__ inline void residual_sum(const __bf16* __restrict__ a,
                                           const __bf16* __restrict__ b,
                                           __bf16* __restrict__ o, int tid) {
  for (int idx = tid; idx < 1024; idx += 256) {
    int r = idx >> 4, c = idx & 15;
    B8 ua, ub, uo;
    ua.q = *(const uint4*)(a + r * WST1 + c * 8);
    ub.q = *(const uint4*)(b + r * WST1 + c * 8);
#pragma unroll
    for (int k = 0; k < 8; ++k) uo.h[k] = (__bf16)((float)ua.h[k] + (float)ub.h[k]);
    *(uint4*)(o + r * WST1 + c * 8) = uo.q;
  }
}

// ---------------------------------------------------------------------------
// Embedding: base = relu(feats @ emb_w^T + emb_b), K=32 -> one WMMA per tile.
// ---------------------------------------------------------------------------
__global__ __launch_bounds__(256, 2) void embed_kernel(
    const float* __restrict__ feats, const __bf16* __restrict__ embW,
    const float* __restrict__ embB,
    __bf16* __restrict__ baseB, __bf16* __restrict__ Ebf, float* __restrict__ outF) {
  extern __shared__ char smem[];
  __bf16* featS = (__bf16*)smem;                                // 64 x 40
  __bf16* wS    = (__bf16*)(smem + 64 * 40 * 2);                // 128 x 40
  __bf16* outS  = (__bf16*)(smem + 64 * 40 * 2 + 128 * 40 * 2); // 64 x WST1
  const int tid = threadIdx.x;
  const int rowBase = blockIdx.x * ROWS;

  {  // stage features fp32 -> bf16
    int r = tid >> 2, c = tid & 3;
    const float* src = feats + (size_t)(rowBase + r) * kFEAT + c * 8;
    B8 o;
#pragma unroll
    for (int k = 0; k < 8; ++k) o.h[k] = (__bf16)src[k];
    *(uint4*)(featS + r * 40 + c * 8) = o.q;
  }
  for (int idx = tid; idx < 512; idx += 256) {
    int r = idx >> 2, c = idx & 3;
    *(uint4*)(wS + r * 40 + c * 8) = *(const uint4*)(embW + r * kFEAT + c * 8);
  }
  __syncthreads();
  {
    const int w = tid >> 5, lane = tid & 31;
    const int rt = w >> 1, ct0 = (w & 1) * 4;
    const int lm = lane & 15, hi = lane >> 4;
    Frag16 a;
    const __bf16* arow = featS + (rt * 16 + lm) * 40;
    a.q[0] = *(const uint4*)(arow + hi * 8);
    a.q[1] = *(const uint4*)(arow + 16 + hi * 8);
#pragma unroll
    for (int j = 0; j < 4; ++j) {
      const int col = (ct0 + j) * 16 + lm;
      const __bf16* wrow = wS + col * 40 + hi * 8;
      Frag16 b;
      b.q[0] = *(const uint4*)(wrow);
      b.q[1] = *(const uint4*)(wrow + 16);
      v8f acc = {};
      acc = __builtin_amdgcn_wmma_f32_16x16x32_bf16(false, a.v, false, b.v,
                                                    (short)0, acc, false, false);
      const float bv = embB[col];
#pragma unroll
      for (int r = 0; r < 8; ++r) {
        float v = acc[r] + bv;
        v = v > 0.f ? v : 0.f;
        outS[(rt * 16 + hi * 8 + r) * WST1 + col] = (__bf16)v;
      }
    }
  }
  __syncthreads();
  const bool inM = rowBase < kM;
  for (int idx = tid; idx < 1024; idx += 256) {
    int r = idx >> 4, c = idx & 15;
    B8 u; u.q = *(const uint4*)(outS + r * WST1 + c * 8);
    size_t go = (size_t)(rowBase + r) * kH + c * 8;
    *(uint4*)(baseB + go) = u.q;
    if (inM) {
      *(uint4*)(Ebf + go) = u.q;
      f32x4 f0 = {(float)u.h[0], (float)u.h[1], (float)u.h[2], (float)u.h[3]};
      f32x4 f1 = {(float)u.h[4], (float)u.h[5], (float)u.h[6], (float)u.h[7]};
      __builtin_nontemporal_store(f0, (f32x4*)(outF + go));
      __builtin_nontemporal_store(f1, (f32x4*)(outF + go + 4));
    }
  }
}

// ---------------------------------------------------------------------------
// Fused 5/6-matmul MLP stack. mode 0: gather-1 (unary), 1: gather-2 (binary),
// 2: concat(base, redux). Activations live in LDS as bf16; weights streamed
// by the Tensor Data Mover, double-buffered (A/B), overlapped with WMMA.
// ---------------------------------------------------------------------------
__global__ __launch_bounds__(256, 2) void mlp_stack_kernel(
    int mode,
    const int* __restrict__ par, const __bf16* __restrict__ gatherE,
    const __bf16* __restrict__ in0, const __bf16* __restrict__ in1,
    const __bf16* __restrict__ w0, const float* __restrict__ b0, int K0,
    const __bf16* __restrict__ w1, const float* __restrict__ b1,
    const __bf16* __restrict__ w2, const float* __restrict__ b2,
    const __bf16* __restrict__ w3, const float* __restrict__ b3,
    const __bf16* __restrict__ w4, const float* __restrict__ b4,
    __bf16* __restrict__ outB, float* __restrict__ outF) {
  extern __shared__ char smem[];
  __bf16* inS  = (__bf16*)smem;                                   // 64 x AST
  __bf16* wSa  = (__bf16*)(smem + ROWS * AST * 2);                // 128 x AST (layers 0,2,4)
  __bf16* wSb  = wSa + 128 * AST;                                 // 128 x WST1 (layers 1,3)
  __bf16* actA = wSb + 128 * WST1;
  __bf16* actB = actA + ROWS * WST1;
  __bf16* actC = actB + ROWS * WST1;
  const int tid = threadIdx.x;
  const int rowBase = blockIdx.x * ROWS;
#if HAVE_TDM
  const bool lead = (tid < 32);  // wave 0 drives the TDM
  const unsigned int offA = (unsigned int)(uintptr_t)wSa;
  const unsigned int offB = (unsigned int)(uintptr_t)wSb;
  if (lead) {                    // prefetch layer-0 and layer-1 weights
    tdm_weight_load(w0, offA, K0);
    tdm_weight_load(w1, offB, kH);
  }
#endif

  // ---- stage input rows into inS (overlaps with TDM weight DMA) ----
  if (mode == 0) {            // unary: gather E[par[r][0]] (128 wide)
    for (int idx = tid; idx < 1024; idx += 256) {
      int r = idx >> 4, c = idx & 15;
      int g = par[2 * (rowBase + r)];
      *(uint4*)(inS + r * AST + c * 8) = *(const uint4*)(gatherE + (size_t)g * kH + c * 8);
    }
  } else if (mode == 1) {     // binary: gather [E[p0] | E[p1]] (256 wide)
    for (int idx = tid; idx < 2048; idx += 256) {
      int r = idx >> 5, c = idx & 31;
      int side = c >> 4, cc = c & 15;
      int g = par[2 * (rowBase + r) + side];
      *(uint4*)(inS + r * AST + c * 8) = *(const uint4*)(gatherE + (size_t)g * kH + cc * 8);
    }
  } else {                    // node: [base[s+r] | redux[r]] (256 wide)
    for (int idx = tid; idx < 2048; idx += 256) {
      int r = idx >> 5, c = idx & 31;
      const __bf16* src = (c < 16)
        ? (in0 + (size_t)(rowBase + r) * kH + c * 8)
        : (in1 + (size_t)(rowBase + r) * kH + (c - 16) * 8);
      *(uint4*)(inS + r * AST + c * 8) = *(const uint4*)src;
    }
  }

#if HAVE_TDM
  if (lead) __builtin_amdgcn_s_wait_tensorcnt(1);     // W0 resident
  __syncthreads();
  mlp_layer(inS, AST, K0, wSa, K0 + 8, b0, actA, tid);            // a0
  __syncthreads();
  if (lead) { tdm_weight_load(w2, offA, kH); __builtin_amdgcn_s_wait_tensorcnt(1); }
  __syncthreads();
  mlp_layer(actA, WST1, kH, wSb, WST1, b1, actB, tid);            // a1
  __syncthreads();
  residual_sum(actA, actB, inS, tid);
  if (lead) { tdm_weight_load(w3, offB, kH); __builtin_amdgcn_s_wait_tensorcnt(1); }
  __syncthreads();
  mlp_layer(inS, WST1, kH, wSa, WST1, b2, actC, tid);             // a2
  __syncthreads();
  residual_sum(actB, actC, inS, tid);
  if (lead) { tdm_weight_load(w4, offA, kH); __builtin_amdgcn_s_wait_tensorcnt(1); }
  __syncthreads();
  mlp_layer(inS, WST1, kH, wSb, WST1, b3, actA, tid);             // a3
  __syncthreads();
  residual_sum(actC, actA, inS, tid);
  if (lead) __builtin_amdgcn_s_wait_tensorcnt(0);
  __syncthreads();
  mlp_layer(inS, WST1, kH, wSa, WST1, b4, actB, tid);             // a4 (final)
  __syncthreads();
#else
  load_weights(w0, wSa, K0, tid);
  __syncthreads();
  mlp_layer(inS, AST, K0, wSa, K0 + 8, b0, actA, tid);
  __syncthreads();
  load_weights(w1, wSb, kH, tid);
  __syncthreads();
  mlp_layer(actA, WST1, kH, wSb, WST1, b1, actB, tid);
  __syncthreads();
  residual_sum(actA, actB, inS, tid);
  load_weights(w2, wSa, kH, tid);
  __syncthreads();
  mlp_layer(inS, WST1, kH, wSa, WST1, b2, actC, tid);
  __syncthreads();
  residual_sum(actB, actC, inS, tid);
  load_weights(w3, wSb, kH, tid);
  __syncthreads();
  mlp_layer(inS, WST1, kH, wSb, WST1, b3, actA, tid);
  __syncthreads();
  residual_sum(actC, actA, inS, tid);
  load_weights(w4, wSa, kH, tid);
  __syncthreads();
  mlp_layer(inS, WST1, kH, wSa, WST1, b4, actB, tid);
  __syncthreads();
#endif

  // ---- write result: bf16 mirror (+ fp32 output, non-temporal) ----
  for (int idx = tid; idx < 1024; idx += 256) {
    int r = idx >> 4, c = idx & 15;
    B8 u; u.q = *(const uint4*)(actB + r * WST1 + c * 8);
    size_t go = (size_t)(rowBase + r) * kH + c * 8;
    *(uint4*)(outB + go) = u.q;
    if (outF) {
      f32x4 f0 = {(float)u.h[0], (float)u.h[1], (float)u.h[2], (float)u.h[3]};
      f32x4 f1 = {(float)u.h[4], (float)u.h[5], (float)u.h[6], (float)u.h[7]};
      __builtin_nontemporal_store(f0, (f32x4*)(outF + go));
      __builtin_nontemporal_store(f1, (f32x4*)(outF + go + 4));
    }
  }
}

// ---------------------------------------------------------------------------
extern "C" void kernel_launch(void* const* d_in, const int* in_sizes, int n_in,
                              void* d_out, int out_size, void* d_ws, size_t ws_size,
                              hipStream_t stream) {
  const float* node_feats = (const float*)d_in[0];
  const float* emb_w  = (const float*)d_in[1];
  const float* emb_b  = (const float*)d_in[2];
  const float* ne0_w  = (const float*)d_in[3];
  const float* ne0_b  = (const float*)d_in[4];
  const float* ne_w   = (const float*)d_in[5];
  const float* ne_b   = (const float*)d_in[6];
  const float* mpu_w  = (const float*)d_in[7];
  const float* mpu_b  = (const float*)d_in[8];
  const float* mpb0_w = (const float*)d_in[9];
  const float* mpb0_b = (const float*)d_in[10];
  const float* mpb_w  = (const float*)d_in[11];
  const float* mpb_b  = (const float*)d_in[12];
  const int*   parents= (const int*)d_in[13];
  float* outE = (float*)d_out;

  char* ws = (char*)d_ws;
  size_t off = 0;
  auto carve = [&](size_t bytes) {
    char* p = ws + off;
    off += (bytes + 255) & ~(size_t)255;
    return p;
  };
  __bf16* Ebf    = (__bf16*)carve((size_t)kN * kH * 2);
  __bf16* baseB  = (__bf16*)carve((size_t)kN * kH * 2);
  __bf16* reduxB = (__bf16*)carve((size_t)kM * kH * 2);
  __bf16* embWB  = (__bf16*)carve((size_t)kH * kFEAT * 2);
  __bf16* ne0WB  = (__bf16*)carve((size_t)kH * 2 * kH * 2);
  __bf16* neWB   = (__bf16*)carve((size_t)4 * kH * kH * 2);
  __bf16* mpuWB  = (__bf16*)carve((size_t)5 * kH * kH * 2);
  __bf16* mpb0WB = (__bf16*)carve((size_t)kH * 2 * kH * 2);
  __bf16* mpbWB  = (__bf16*)carve((size_t)4 * kH * kH * 2);

  auto cvt = [&](const float* s, __bf16* d, int n) {
    cvt_f32_bf16<<<(n + 255) / 256, 256, 0, stream>>>(s, d, n);
  };
  cvt(emb_w,  embWB,  kH * kFEAT);
  cvt(ne0_w,  ne0WB,  kH * 2 * kH);
  cvt(ne_w,   neWB,   4 * kH * kH);
  cvt(mpu_w,  mpuWB,  5 * kH * kH);
  cvt(mpb0_w, mpb0WB, kH * 2 * kH);
  cvt(mpb_w,  mpbWB,  4 * kH * kH);

  const size_t smemE = (size_t)(64 * 40 + 128 * 40 + 64 * WST1) * 2;
  embed_kernel<<<(int)(kN / ROWS), 256, smemE, stream>>>(node_feats, embWB, emb_b,
                                                         baseB, Ebf, outE);

  const size_t smemM =
      (size_t)(ROWS * AST + 128 * AST + 128 * WST1 + 3 * ROWS * WST1) * 2;
  for (int l = 1; l < kL; ++l) {
    size_t s = (size_t)l * kM;
    // unary redux (rows [0, MU))
    mlp_stack_kernel<<<kMU / ROWS, 256, smemM, stream>>>(
        0, parents + 2 * s, Ebf, nullptr, nullptr,
        mpuWB,               mpu_b,          kH,
        mpuWB + 1 * kH * kH, mpu_b + 1 * kH,
        mpuWB + 2 * kH * kH, mpu_b + 2 * kH,
        mpuWB + 3 * kH * kH, mpu_b + 3 * kH,
        mpuWB + 4 * kH * kH, mpu_b + 4 * kH,
        reduxB, nullptr);
    // binary redux (rows [MU, M))
    mlp_stack_kernel<<<(kM - kMU) / ROWS, 256, smemM, stream>>>(
        1, parents + 2 * (s + kMU), Ebf, nullptr, nullptr,
        mpb0WB,              mpb0_b,         2 * kH,
        mpbWB + 0 * kH * kH, mpb_b + 0 * kH,
        mpbWB + 1 * kH * kH, mpb_b + 1 * kH,
        mpbWB + 2 * kH * kH, mpb_b + 2 * kH,
        mpbWB + 3 * kH * kH, mpb_b + 3 * kH,
        reduxB + (size_t)kMU * kH, nullptr);
    // node embed (all M rows) -> E[s:s+M] (bf16 mirror + fp32 out)
    mlp_stack_kernel<<<kM / ROWS, 256, smemM, stream>>>(
        2, nullptr, nullptr, baseB + s * kH, reduxB,
        ne0WB,              ne0_b,          2 * kH,
        neWB + 0 * kH * kH, ne_b + 0 * kH,
        neWB + 1 * kH * kH, ne_b + 1 * kH,
        neWB + 2 * kH * kH, ne_b + 2 * kH,
        neWB + 3 * kH * kH, ne_b + 3 * kH,
        Ebf + s * kH, outE + s * kH);
  }
}
